// AGCN_52853867544726
// MI455X (gfx1250) — matrix operand order repaired and checked by reading, not compiled, not run on hardware
//
#include <hip/hip_runtime.h>
#include <math.h>

// ---------------------------------------------------------------------------
// Types for CDNA5 WMMA (wave32, 16x16x32 bf16 -> f32)
// ---------------------------------------------------------------------------
typedef __attribute__((ext_vector_type(16))) __bf16 v16bf;
typedef __attribute__((ext_vector_type(8)))  float  v8f;
typedef __attribute__((ext_vector_type(8)))  __bf16 bf16x8;

#define LEAKY_SLOPE 0.01f

__device__ __forceinline__ float apply_act(float v, int act) {
  if (act == 1) return fmaxf(v, 0.f);                 // relu
  if (act == 2) return v > 0.f ? v : LEAKY_SLOPE * v; // leaky_relu
  return v;
}

__device__ __forceinline__ float wave_sum(float v) {
  #pragma unroll
  for (int off = 16; off > 0; off >>= 1) v += __shfl_down(v, off, 32);
  return v;
}

// CDNA5 async global->LDS copy (ASYNCcnt-tracked). LDS generic addresses keep
// the byte offset in the low 32 bits, so truncation yields the DS address.
__device__ __forceinline__ void async_copy_b128(void* lds_dst, const void* gsrc) {
  unsigned ldsoff = (unsigned)(uintptr_t)lds_dst;
  unsigned long long ga = (unsigned long long)(uintptr_t)gsrc;
  asm volatile("global_load_async_to_lds_b128 %0, %1, off"
               :: "v"(ldsoff), "v"(ga) : "memory");
}
__device__ __forceinline__ void wait_async0() {
  asm volatile("s_wait_asynccnt 0x0" ::: "memory");
}

// ---------------------------------------------------------------------------
// Weight pre-pass: WT[o][k] = (bf16)W[k][o], zero-padded to [Opad][Kpad].
// Weights are tiny (<=16MB) and L2-resident; this removes all bounds checks
// and conversions from the GEMM hot loop.
// ---------------------------------------------------------------------------
__global__ void agcn_wT(const float* __restrict__ W, __bf16* __restrict__ WT,
                        int K, int O, int Kpad, int Opad) {
  int idx = blockIdx.x * blockDim.x + threadIdx.x;
  if (idx >= Opad * Kpad) return;
  int o = idx / Kpad, k = idx % Kpad;
  WT[idx] = (o < O && k < K) ? (__bf16)W[(size_t)k * O + o] : (__bf16)0.f;
}

// ---------------------------------------------------------------------------
// WMMA GEMM: C[N,O] = act(A[N,K] @ W[K,O] + bias), B pre-transposed bf16.
// Block tile 64(M) x 128(N), 8 waves, wave computes 32x32 via 4 WMMA accs.
// Double-buffered LDS; B tiles arrive via global_load_async_to_lds_b128.
// ---------------------------------------------------------------------------
#define BM 64
#define BN 128
#define KT 32

__global__ __launch_bounds__(256)
void agcn_gemm_wmma(const float* __restrict__ A, const __bf16* __restrict__ WT,
                    const float* __restrict__ bias, float* __restrict__ C,
                    int Nrows, int Kdim, int Ocols, int Kpad, int act) {
  __shared__ __bf16 lA[2][BM][KT];   // 2 x 4 KB
  __shared__ __bf16 lB[2][BN][KT];   // 2 x 8 KB  ([n][k])

  const int tid  = threadIdx.x;
  const int wave = tid >> 5;
  const int lane = tid & 31;
  const int wm   = wave >> 2;        // 0..1 -> 32-row half
  const int wn   = wave & 3;         // 0..3 -> 32-col group
  const int blockM = blockIdx.y * BM;
  const int blockN = blockIdx.x * BN;

  // --- staging helpers -----------------------------------------------------
  auto stageA = [&](int buf, int k0) {
    if ((blockM + BM <= Nrows) && (k0 + KT <= Kdim)) {
      // interior: branch-free, 2x float4 load + one 16B LDS store per thread
      const int m = tid >> 2;               // 0..63
      const int k = (tid & 3) * 8;          // 0,8,16,24
      const float* ap = A + (size_t)(blockM + m) * Kdim + k0 + k;
      float4 va = *(const float4*)ap;
      float4 vb = *(const float4*)(ap + 4);
      bf16x8 h;
      h[0] = (__bf16)va.x; h[1] = (__bf16)va.y; h[2] = (__bf16)va.z; h[3] = (__bf16)va.w;
      h[4] = (__bf16)vb.x; h[5] = (__bf16)vb.y; h[6] = (__bf16)vb.z; h[7] = (__bf16)vb.w;
      *(bf16x8*)&lA[buf][m][k] = h;
    } else {
      for (int i = tid; i < BM * KT; i += 256) {
        int m = i >> 5, k = i & 31;
        int gm = blockM + m, gk = k0 + k;
        float v = (gm < Nrows && gk < Kdim) ? A[(size_t)gm * Kdim + gk] : 0.f;
        lA[buf][m][k] = (__bf16)v;
      }
    }
  };
  auto stageB = [&](int buf, int k0) {
    // pure byte copy from padded bf16 W^T: async DMA into LDS, no VGPR data
    #pragma unroll
    for (int t = 0; t < 2; ++t) {
      int idx = tid + t * 256;              // 0..511 -> 512 x 16B = 8 KB
      int n = idx >> 2;                     // 0..127
      int k = (idx & 3) * 8;                // 0,8,16,24
      async_copy_b128(&lB[buf][n][k],
                      WT + (size_t)(blockN + n) * Kpad + k0 + k);
    }
  };

  v8f acc[2][2];
  #pragma unroll
  for (int mi = 0; mi < 2; ++mi)
    #pragma unroll
    for (int ni = 0; ni < 2; ++ni)
      #pragma unroll
      for (int r = 0; r < 8; ++r) acc[mi][ni][r] = 0.f;

  const int kSteps = (Kdim + KT - 1) / KT;
  stageA(0, 0);
  stageB(0, 0);

  for (int ks = 0; ks < kSteps; ++ks) {
    const int cur = ks & 1;
    wait_async0();            // this wave's async writes into lB[cur] landed
    __syncthreads();          // everyone's tile visible; prev reads done

    if (ks + 1 < kSteps) {    // overlap next-tile DMA with this tile's math
      stageA(cur ^ 1, (ks + 1) * KT);
      stageB(cur ^ 1, (ks + 1) * KT);
    }

    // A fragment: 16x32 bf16 per 16-row tile (ISA 7.12.2 layout).
    v16bf af[2], bfb[2];
    #pragma unroll
    for (int mi = 0; mi < 2; ++mi) {
      const int m  = wm * 32 + mi * 16 + (lane & 15);
      const int kb = (lane >> 4) * 8;       // lanes 16..31 -> K offset 8
      #pragma unroll
      for (int p = 0; p < 8; ++p) {
        const int kk = ((p & 4) ? 16 : 0) + kb + (p & 3) * 2;
        af[mi][2 * p]     = lA[cur][m][kk];
        af[mi][2 * p + 1] = lA[cur][m][kk + 1];
      }
    }
    // B fragment: 32x16 bf16, lane%16 = column, lanes 16..31 -> K offset 8.
    #pragma unroll
    for (int ni = 0; ni < 2; ++ni) {
      const int n  = wn * 32 + ni * 16 + (lane & 15);
      const int kb = (lane >> 4) * 8;
      #pragma unroll
      for (int p = 0; p < 8; ++p) {
        const int kk = ((p & 4) ? 16 : 0) + kb + (p & 3) * 2;
        bfb[ni][2 * p]     = lB[cur][n][kk];
        bfb[ni][2 * p + 1] = lB[cur][n][kk + 1];
      }
    }
    #pragma unroll
    for (int mi = 0; mi < 2; ++mi)
      #pragma unroll
      for (int ni = 0; ni < 2; ++ni)
        acc[mi][ni] = __builtin_amdgcn_wmma_f32_16x16x32_bf16(
            false, af[mi], false, bfb[ni], (short)0, acc[mi][ni], false, false);
  }
  wait_async0();

  // Store: C/D layout -> n = lane&15, m = r + 8*(lane>=16).
  #pragma unroll
  for (int mi = 0; mi < 2; ++mi) {
    #pragma unroll
    for (int ni = 0; ni < 2; ++ni) {
      const int n = blockN + wn * 32 + ni * 16 + (lane & 15);
      if (n >= Ocols) continue;
      const float bv = bias ? bias[n] : 0.f;
      #pragma unroll
      for (int r = 0; r < 8; ++r) {
        const int m = blockM + wm * 32 + mi * 16 + r + ((lane >> 4) << 3);
        if (m < Nrows)
          C[(size_t)m * Ocols + n] = apply_act(acc[mi][ni][r] + bv, act);
      }
    }
  }
}

// ---------------------------------------------------------------------------
// Per-thread rowdot for small-K matmul (dh1: K=10 -> O=2000): coalesced over O.
// ---------------------------------------------------------------------------
__global__ __launch_bounds__(256)
void agcn_rowdot(const float* __restrict__ A, const float* __restrict__ W,
                 const float* __restrict__ bias, float* __restrict__ C,
                 int Nrows, int Kdim, int Ocols, int act) {
  int idx = blockIdx.x * blockDim.x + threadIdx.x;
  if (idx >= Nrows * Ocols) return;
  int n = idx / Ocols, o = idx % Ocols;
  float s = bias ? bias[o] : 0.f;
  const float* a = A + (size_t)n * Kdim;
  for (int k = 0; k < Kdim; ++k) s += a[k] * W[(size_t)k * Ocols + o];
  C[idx] = apply_act(s, act);
}

// Wave-per-row dot for O=10 with large K (z, t4): lane-strided K, coalesced.
__global__ __launch_bounds__(256)
void agcn_rowdot10(const float* __restrict__ A, const float* __restrict__ W,
                   const float* __restrict__ bias, float* __restrict__ C,
                   int Nrows, int Kdim) {
  int node = blockIdx.x * 8 + (threadIdx.x >> 5);
  int lane = threadIdx.x & 31;
  if (node >= Nrows) return;
  float acc[10];
  #pragma unroll
  for (int j = 0; j < 10; ++j) acc[j] = 0.f;
  const float* a = A + (size_t)node * Kdim;
  for (int k = lane; k < Kdim; k += 32) {
    float v = a[k];
    const float* wr = W + (size_t)k * 10;
    #pragma unroll
    for (int j = 0; j < 10; ++j) acc[j] += v * wr[j];
  }
  #pragma unroll
  for (int j = 0; j < 10; ++j) acc[j] = wave_sum(acc[j]);
  if (lane == 0) {
    #pragma unroll
    for (int j = 0; j < 10; ++j)
      C[(size_t)node * 10 + j] = acc[j] + (bias ? bias[j] : 0.f);
  }
}

// ---------------------------------------------------------------------------
// Degree / normalization
// ---------------------------------------------------------------------------
__global__ void agcn_zero(float* __restrict__ p, size_t count) {
  size_t i = (size_t)blockIdx.x * blockDim.x + threadIdx.x;
  if (i < count) p[i] = 0.f;
}

__global__ void agcn_degcount(const int* __restrict__ src, const int* __restrict__ dst,
                              float* __restrict__ dout, float* __restrict__ din, int E) {
  int e = blockIdx.x * blockDim.x + threadIdx.x;
  if (e >= E) return;
  atomicAdd(&dout[src[e]], 1.f);
  atomicAdd(&din[dst[e]], 1.f);
}

__global__ void agcn_norm(float* __restrict__ dout, float* __restrict__ din, int N) {
  int n = blockIdx.x * blockDim.x + threadIdx.x;
  if (n >= N) return;
  dout[n] = rsqrtf(dout[n] + 1.f);  // +1 self-loop; deg>=1 always
  din[n]  = rsqrtf(din[n] + 1.f);
}

// ---------------------------------------------------------------------------
// Edge scatter: agg[d] += t[s] * nsrc[s]  (self loops appended logically)
// ---------------------------------------------------------------------------
__global__ __launch_bounds__(256)
void agcn_scatter(const float* __restrict__ t, float* __restrict__ agg,
                  const int* __restrict__ src, const int* __restrict__ dst,
                  const float* __restrict__ nsrc, int E, int N, int Wd) {
  int e = blockIdx.x * 8 + (threadIdx.x >> 5);
  int lane = threadIdx.x & 31;
  if (e >= E + N) return;
  int s, d;
  if (e < E) { s = src[e]; d = dst[e]; } else { s = e - E; d = s; }
  const float ns = nsrc[s];
  const float* tr = t + (size_t)s * Wd;
  float* ag = agg + (size_t)d * Wd;
  for (int f = lane; f < Wd; f += 32) atomicAdd(&ag[f], tr[f] * ns);
}

__global__ void agcn_rowscale(const float* __restrict__ in, const float* __restrict__ ndst,
                              float* __restrict__ out, int N, int Wd, int act) {
  int idx = blockIdx.x * blockDim.x + threadIdx.x;
  if (idx >= N * Wd) return;
  int n = idx / Wd;
  out[idx] = apply_act(in[idx] * ndst[n], act);
}

// ---------------------------------------------------------------------------
// Gate: m = l2norm(softmax(leaky([tra, z] @ w + b)))  (2 outputs)
// ---------------------------------------------------------------------------
__global__ __launch_bounds__(256)
void agcn_gate(const float* __restrict__ ta, const float* __restrict__ zb,
               const float* __restrict__ w, const float* __restrict__ b,
               float* __restrict__ m, int N, int H) {
  int node = blockIdx.x * 8 + (threadIdx.x >> 5);
  int lane = threadIdx.x & 31;
  if (node >= N) return;
  float a0 = 0.f, a1 = 0.f;
  const float* ar = ta + (size_t)node * H;
  const float* zr = zb + (size_t)node * H;
  for (int k = lane; k < H; k += 32) {
    float av = ar[k], zv = zr[k];
    a0 += av * w[k * 2 + 0] + zv * w[(H + k) * 2 + 0];
    a1 += av * w[k * 2 + 1] + zv * w[(H + k) * 2 + 1];
  }
  a0 = wave_sum(a0); a1 = wave_sum(a1);
  if (lane == 0) {
    float l0 = a0 + b[0], l1 = a1 + b[1];
    l0 = l0 > 0.f ? l0 : LEAKY_SLOPE * l0;
    l1 = l1 > 0.f ? l1 : LEAKY_SLOPE * l1;
    float mx = fmaxf(l0, l1);
    float e0 = expf(l0 - mx), e1 = expf(l1 - mx);
    float s = e0 + e1;
    float p0 = e0 / s, p1 = e1 / s;
    float nrm = fmaxf(sqrtf(p0 * p0 + p1 * p1), 1e-12f);
    m[node * 2 + 0] = p0 / nrm;
    m[node * 2 + 1] = p1 / nrm;
  }
}

__global__ void agcn_mix(const float* __restrict__ m, const float* __restrict__ zi,
                         const float* __restrict__ tr, float* __restrict__ out,
                         int N, int Wd) {
  int idx = blockIdx.x * blockDim.x + threadIdx.x;
  if (idx >= N * Wd) return;
  int n = idx / Wd;
  out[idx] = m[n * 2 + 0] * zi[idx] + m[n * 2 + 1] * tr[idx];
}

// ---------------------------------------------------------------------------
// Fused segment reader: concat [z1(500), z2(500), z3(2000), z4(10), z(10)]
// ---------------------------------------------------------------------------
__device__ __forceinline__ float seg_val(int node, int idx,
                                         const float* z1, const float* z2,
                                         const float* z3, const float* z4,
                                         const float* z, int* seg) {
  if (idx < 500)  { *seg = 0; return z1[(size_t)node * 500 + idx]; }
  if (idx < 1000) { *seg = 1; return z2[(size_t)node * 500 + idx - 500]; }
  if (idx < 3000) { *seg = 2; return z3[(size_t)node * 2000 + idx - 1000]; }
  if (idx < 3010) { *seg = 3; return z4[(size_t)node * 10 + idx - 3000]; }
  *seg = 4; return z[(size_t)node * 10 + idx - 3010];
}

__global__ __launch_bounds__(256)
void agcn_u(const float* __restrict__ z1, const float* __restrict__ z2,
            const float* __restrict__ z3, const float* __restrict__ z4,
            const float* __restrict__ z, const float* __restrict__ mlw,
            const float* __restrict__ mlb, float* __restrict__ u, int N) {
  int node = blockIdx.x * 8 + (threadIdx.x >> 5);
  int lane = threadIdx.x & 31;
  if (node >= N) return;
  float acc[5] = {0.f, 0.f, 0.f, 0.f, 0.f};
  for (int idx = lane; idx < 3020; idx += 32) {
    int seg; float v = seg_val(node, idx, z1, z2, z3, z4, z, &seg);
    const float* wr = mlw + (size_t)idx * 5;
    #pragma unroll
    for (int j = 0; j < 5; ++j) acc[j] += v * wr[j];
  }
  #pragma unroll
  for (int j = 0; j < 5; ++j) acc[j] = wave_sum(acc[j]);
  if (lane == 0) {
    float l[5], mx = -1e30f;
    #pragma unroll
    for (int j = 0; j < 5; ++j) {
      float t = acc[j] + mlb[j];
      l[j] = t > 0.f ? t : LEAKY_SLOPE * t;
      mx = fmaxf(mx, l[j]);
    }
    float s = 0.f;
    #pragma unroll
    for (int j = 0; j < 5; ++j) { l[j] = expf(l[j] - mx); s += l[j]; }
    float nrm = 0.f;
    #pragma unroll
    for (int j = 0; j < 5; ++j) { l[j] /= s; nrm += l[j] * l[j]; }
    nrm = fmaxf(sqrtf(nrm), 1e-12f);
    #pragma unroll
    for (int j = 0; j < 5; ++j) u[(size_t)node * 5 + j] = l[j] / nrm;
  }
}

__global__ __launch_bounds__(256)
void agcn_netg5(const float* __restrict__ z1, const float* __restrict__ z2,
                const float* __restrict__ z3, const float* __restrict__ z4,
                const float* __restrict__ z, const float* __restrict__ u,
                const float* __restrict__ g5, float* __restrict__ t5, int N) {
  int node = blockIdx.x * 8 + (threadIdx.x >> 5);
  int lane = threadIdx.x & 31;
  if (node >= N) return;
  float uw[5];
  #pragma unroll
  for (int j = 0; j < 5; ++j) uw[j] = u[(size_t)node * 5 + j];
  float acc[10];
  #pragma unroll
  for (int j = 0; j < 10; ++j) acc[j] = 0.f;
  for (int idx = lane; idx < 3020; idx += 32) {
    int seg; float v = seg_val(node, idx, z1, z2, z3, z4, z, &seg);
    v *= uw[seg];
    const float* wr = g5 + (size_t)idx * 10;
    #pragma unroll
    for (int j = 0; j < 10; ++j) acc[j] += v * wr[j];
  }
  #pragma unroll
  for (int j = 0; j < 10; ++j) acc[j] = wave_sum(acc[j]);
  if (lane == 0) {
    #pragma unroll
    for (int j = 0; j < 10; ++j) t5[(size_t)node * 10 + j] = acc[j];
  }
}

// ---------------------------------------------------------------------------
// predict = softmax(h); Student-t q/p
// ---------------------------------------------------------------------------
__global__ void agcn_softmax10(const float* __restrict__ h, float* __restrict__ out, int N) {
  int n = blockIdx.x * blockDim.x + threadIdx.x;
  if (n >= N) return;
  float v[10], mx = -1e30f;
  #pragma unroll
  for (int j = 0; j < 10; ++j) { v[j] = h[(size_t)n * 10 + j]; mx = fmaxf(mx, v[j]); }
  float s = 0.f;
  #pragma unroll
  for (int j = 0; j < 10; ++j) { v[j] = expf(v[j] - mx); s += v[j]; }
  #pragma unroll
  for (int j = 0; j < 10; ++j) out[(size_t)n * 10 + j] = v[j] / s;
}

__global__ void agcn_q(const float* __restrict__ z, const float* __restrict__ cluster,
                       float* __restrict__ qout, float* __restrict__ colsum, int N) {
  int n = blockIdx.x * blockDim.x + threadIdx.x;
  if (n >= N) return;
  float zr[10];
  #pragma unroll
  for (int j = 0; j < 10; ++j) zr[j] = z[(size_t)n * 10 + j];
  float qv[10], s = 0.f;
  #pragma unroll
  for (int k = 0; k < 10; ++k) {
    float dist = 0.f;
    #pragma unroll
    for (int j = 0; j < 10; ++j) {
      float d = zr[j] - cluster[k * 10 + j];
      dist += d * d;
    }
    qv[k] = 1.f / (1.f + dist);   // V=1 -> exponent (V+1)/2 = 1
    s += qv[k];
  }
  #pragma unroll
  for (int k = 0; k < 10; ++k) {
    float q = qv[k] / s;
    qout[(size_t)n * 10 + k] = q;
    atomicAdd(&colsum[k], q);
  }
}

__global__ void agcn_p(const float* __restrict__ qin, const float* __restrict__ colsum,
                       float* __restrict__ pout, int N) {
  int n = blockIdx.x * blockDim.x + threadIdx.x;
  if (n >= N) return;
  float w[10], s = 0.f;
  #pragma unroll
  for (int k = 0; k < 10; ++k) {
    float q = qin[(size_t)n * 10 + k];
    w[k] = q * q / colsum[k];
    s += w[k];
  }
  #pragma unroll
  for (int k = 0; k < 10; ++k) pout[(size_t)n * 10 + k] = w[k] / s;
}

// ---------------------------------------------------------------------------
// Host orchestration
// ---------------------------------------------------------------------------
static inline void launch_gemm(const float* A, const float* W, const float* bias,
                               float* C, __bf16* WT, int Nr, int K, int O, int act,
                               hipStream_t s) {
  const int Kpad = ((K + KT - 1) / KT) * KT;
  const int Opad = ((O + BN - 1) / BN) * BN;
  const int total = Opad * Kpad;
  agcn_wT<<<(total + 255) / 256, 256, 0, s>>>(W, WT, K, O, Kpad, Opad);
  dim3 grid(Opad / BN, (Nr + BM - 1) / BM);
  agcn_gemm_wmma<<<grid, 256, 0, s>>>(A, WT, bias, C, Nr, K, O, Kpad, act);
}
static inline void launch_rowdot(const float* A, const float* W, const float* bias,
                                 float* C, int Nr, int K, int O, int act, hipStream_t s) {
  int total = Nr * O;
  agcn_rowdot<<<(total + 255) / 256, 256, 0, s>>>(A, W, bias, C, Nr, K, O, act);
}
static inline void launch_zero(float* p, size_t count, hipStream_t s) {
  agcn_zero<<<(unsigned)((count + 255) / 256), 256, 0, s>>>(p, count);
}

extern "C" void kernel_launch(void* const* d_in, const int* in_sizes, int n_in,
                              void* d_out, int out_size, void* d_ws, size_t ws_size,
                              hipStream_t stream) {
  (void)n_in; (void)out_size; (void)ws_size;
  const int NI = 2000, H1 = 500, H2 = 500, H3 = 2000, NZ = 10;
  const int N = in_sizes[0] / NI;      // 20000
  const int E = in_sizes[31];          // 320000

  const float* x   = (const float*)d_in[0];
  const float* ew1 = (const float*)d_in[1];  const float* eb1 = (const float*)d_in[2];
  const float* ew2 = (const float*)d_in[3];  const float* eb2 = (const float*)d_in[4];
  const float* ew3 = (const float*)d_in[5];  const float* eb3 = (const float*)d_in[6];
  const float* zw  = (const float*)d_in[7];  const float* zb  = (const float*)d_in[8];
  const float* dw1 = (const float*)d_in[9];  const float* db1 = (const float*)d_in[10];
  const float* dw2 = (const float*)d_in[11]; const float* db2 = (const float*)d_in[12];
  const float* dw3 = (const float*)d_in[13]; const float* db3 = (const float*)d_in[14];
  const float* xw  = (const float*)d_in[15]; const float* xb  = (const float*)d_in[16];
  const float* g1  = (const float*)d_in[17];
  const float* g2  = (const float*)d_in[18];
  const float* g3  = (const float*)d_in[19];
  const float* g4  = (const float*)d_in[20];
  const float* g5  = (const float*)d_in[21];
  const float* m1w = (const float*)d_in[22]; const float* m1b = (const float*)d_in[23];
  const float* m2w = (const float*)d_in[24]; const float* m2b = (const float*)d_in[25];
  const float* m3w = (const float*)d_in[26]; const float* m3b = (const float*)d_in[27];
  const float* mlw = (const float*)d_in[28]; const float* mlb = (const float*)d_in[29];
  const float* cluster = (const float*)d_in[30];
  const int* src = (const int*)d_in[31];
  const int* dst = (const int*)d_in[32];

  float* out = (float*)d_out;
  float* out_xbar = out;
  float* out_q    = out + (size_t)N * NI;
  float* out_pred = out_q + (size_t)N * NZ;
  float* out_p    = out_pred + (size_t)N * NZ;

  // --- bump allocator on d_ws ---
  char* wsb = (char*)d_ws;
  size_t off = 0;
  auto alloc = [&](size_t bytes) -> void* {
    void* p = (void*)(wsb + off);
    off = (off + bytes + 255) & ~(size_t)255;
    return p;
  };
  float* nsrc = (float*)alloc((size_t)N * 4);
  float* ndst = (float*)alloc((size_t)N * 4);
  float* tra1 = (float*)alloc((size_t)N * H1 * 4);
  float* tra2 = (float*)alloc((size_t)N * H2 * 4);
  float* tra3 = (float*)alloc((size_t)N * H3 * 4);
  float* zbuf = (float*)alloc((size_t)N * NZ * 4);
  float* z1   = (float*)alloc((size_t)N * H1 * 4);
  float* z2   = (float*)alloc((size_t)N * H2 * 4);
  float* z3   = (float*)alloc((size_t)N * H3 * 4);
  float* z4   = (float*)alloc((size_t)N * NZ * 4);
  float* bigA = (float*)alloc((size_t)N * H3 * 4);   // 160 MB scratch
  float* bigB = (float*)alloc((size_t)N * H3 * 4);   // 160 MB scratch
  float* mbuf = (float*)alloc((size_t)N * 2 * 4);
  float* ubuf = (float*)alloc((size_t)N * 5 * 4);
  float* s10a = (float*)alloc((size_t)N * NZ * 4);
  float* s10b = (float*)alloc((size_t)N * NZ * 4);
  float* hbuf = (float*)alloc((size_t)N * NZ * 4);
  float* colsum = (float*)alloc(64);
  __bf16* wtb = (__bf16*)alloc((size_t)2048 * 2048 * 2);  // padded bf16 W^T

  const int edgesTot = E + N;
  const dim3 edgeGrid((edgesTot + 7) / 8);
  const dim3 waveGrid((N + 7) / 8);

  // 1) degrees -> symmetric norms (self-loops folded in as +1)
  launch_zero(nsrc, (size_t)N, stream);
  launch_zero(ndst, (size_t)N, stream);
  agcn_degcount<<<(E + 255) / 256, 256, 0, stream>>>(src, dst, nsrc, ndst, E);
  agcn_norm<<<(N + 255) / 256, 256, 0, stream>>>(nsrc, ndst, N);

  // 2) autoencoder (WMMA for the fat GEMMs)
  launch_gemm(x,    ew1, eb1, tra1, wtb, N, NI, H1, 1, stream);
  launch_gemm(tra1, ew2, eb2, tra2, wtb, N, H1, H2, 1, stream);
  launch_gemm(tra2, ew3, eb3, tra3, wtb, N, H2, H3, 1, stream);
  agcn_rowdot10<<<waveGrid, 256, 0, stream>>>(tra3, zw, zb, zbuf, N, H3);
  launch_rowdot(zbuf, dw1, db1, bigA, N, NZ, H3, 1, stream);        // dh1
  launch_gemm(bigA, dw2, db2, bigB, wtb, N, H3, H2, 1, stream);     // dh2
  launch_gemm(bigB, dw3, db3, bigA, wtb, N, H2, H1, 1, stream);     // dh3
  launch_gemm(bigA, xw,  xb,  out_xbar, wtb, N, H1, NI, 0, stream); // x_bar

  // 3) z1 = gconv(x, g1)
  launch_gemm(x, g1, nullptr, bigA, wtb, N, NI, H1, 0, stream);
  launch_zero(bigB, (size_t)N * H1, stream);
  agcn_scatter<<<edgeGrid, 256, 0, stream>>>(bigA, bigB, src, dst, nsrc, E, N, H1);
  agcn_rowscale<<<(N * H1 + 255) / 256, 256, 0, stream>>>(bigB, ndst, z1, N, H1, 2);

  // 4) m1, mix1, z2 = gconv(mix1, g2)
  agcn_gate<<<waveGrid, 256, 0, stream>>>(tra1, z1, m1w, m1b, mbuf, N, H1);
  agcn_mix<<<(N * H1 + 255) / 256, 256, 0, stream>>>(mbuf, z1, tra1, bigA, N, H1);
  launch_zero(bigB, (size_t)N * H1, stream);
  agcn_scatter<<<edgeGrid, 256, 0, stream>>>(bigA, bigB, src, dst, nsrc, E, N, H1);
  agcn_rowscale<<<(N * H1 + 255) / 256, 256, 0, stream>>>(bigB, ndst, bigB, N, H1, 0);
  launch_gemm(bigB, g2, nullptr, z2, wtb, N, H1, H2, 2, stream);

  // 5) m2, mix2, z3 = gconv(mix2, g3)
  agcn_gate<<<waveGrid, 256, 0, stream>>>(tra2, z2, m2w, m2b, mbuf, N, H2);
  agcn_mix<<<(N * H2 + 255) / 256, 256, 0, stream>>>(mbuf, z2, tra2, bigA, N, H2);
  launch_zero(bigB, (size_t)N * H2, stream);
  agcn_scatter<<<edgeGrid, 256, 0, stream>>>(bigA, bigB, src, dst, nsrc, E, N, H2);
  agcn_rowscale<<<(N * H2 + 255) / 256, 256, 0, stream>>>(bigB, ndst, bigB, N, H2, 0);
  launch_gemm(bigB, g3, nullptr, z3, wtb, N, H2, H3, 2, stream);

  // 6) m3, mix3, z4 = gconv(mix3, g4)
  agcn_gate<<<waveGrid, 256, 0, stream>>>(tra3, z3, m3w, m3b, mbuf, N, H3);
  agcn_mix<<<(N * H3 + 255) / 256, 256, 0, stream>>>(mbuf, z3, tra3, bigA, N, H3);
  agcn_rowdot10<<<waveGrid, 256, 0, stream>>>(bigA, g4, nullptr, s10a, N, H3);
  launch_zero(s10b, (size_t)N * NZ, stream);
  agcn_scatter<<<edgeGrid, 256, 0, stream>>>(s10a, s10b, src, dst, nsrc, E, N, NZ);
  agcn_rowscale<<<(N * NZ + 255) / 256, 256, 0, stream>>>(s10b, ndst, z4, N, NZ, 2);

  // 7) attention u over the 3020-wide concat, then fused net@g5 + gconv
  agcn_u<<<waveGrid, 256, 0, stream>>>(z1, z2, z3, z4, zbuf, mlw, mlb, ubuf, N);
  agcn_netg5<<<waveGrid, 256, 0, stream>>>(z1, z2, z3, z4, zbuf, ubuf, g5, s10a, N);
  launch_zero(s10b, (size_t)N * NZ, stream);
  agcn_scatter<<<edgeGrid, 256, 0, stream>>>(s10a, s10b, src, dst, nsrc, E, N, NZ);
  agcn_rowscale<<<(N * NZ + 255) / 256, 256, 0, stream>>>(s10b, ndst, hbuf, N, NZ, 0);
  agcn_softmax10<<<(N + 255) / 256, 256, 0, stream>>>(hbuf, out_pred, N);

  // 8) Student-t q and target p
  launch_zero(colsum, 16, stream);
  agcn_q<<<(N + 255) / 256, 256, 0, stream>>>(zbuf, cluster, out_q, colsum, N);
  agcn_p<<<(N + 255) / 256, 256, 0, stream>>>(out_q, colsum, out_p, N);
}
